// MTLModel_68461778698597
// MI455X (gfx1250) — compile-verified, matrix operand order checked by reading, not run on hardware
//
#include <hip/hip_runtime.h>
#include <hip/hip_bf16.h>
#include <cstdint>

// Problem constants (match reference)
#define BQ    256
#define SSUBQ 192
#define SEQQ  128
#define DQ    768
#define HQ    256
#define GQ    1024          // 4*H
#define MQ    (BQ*SEQQ)     // 32768 rows for the big GEMMs

typedef __bf16   bf16_t;
typedef _Float16 f16_t;
typedef __attribute__((ext_vector_type(16))) __bf16 bf16x16;
typedef __attribute__((ext_vector_type(8)))  __bf16 bf16x8;
typedef __attribute__((ext_vector_type(8)))  float  f32x8;

__device__ __forceinline__ float sigf(float x) { return 1.0f / (1.0f + expf(-x)); }

// ---------------------------------------------------------------------------
// CDNA5 async global->LDS copy (ASYNCcnt-tracked) + wait
// ---------------------------------------------------------------------------
__device__ __forceinline__ uint32_t lds_off(const void* p) {
    // addrspace(3) flat form keeps the LDS byte offset in the low 32 bits
    return (uint32_t)(uintptr_t)p;
}
__device__ __forceinline__ void async_b128(uint32_t lds, const void* g) {
    asm volatile("global_load_async_to_lds_b128 %0, %1, off" :: "v"(lds), "v"(g) : "memory");
}
__device__ __forceinline__ void wait_async0() {
    asm volatile("s_wait_asynccnt 0" ::: "memory");
}

// ---------------------------------------------------------------------------
// WMMA helpers (16-bit A/B layout, CDNA5 ISA 7.12.2): lane l (g=l>>4, r=l&15)
// holds two contiguous 8-element K-runs: [8g,8g+8) and [16+8g,16+8g+8).
// ---------------------------------------------------------------------------
__device__ __forceinline__ bf16x16 load_frag(const bf16_t* rowptr) {
    bf16x16 v;
    ((bf16x8*)&v)[0] = *(const bf16x8*)(rowptr);
    ((bf16x8*)&v)[1] = *(const bf16x8*)(rowptr + 16);
    return v;
}
__device__ __forceinline__ f32x8 bwmma(bf16x16 a, bf16x16 b, f32x8 c) {
    return __builtin_amdgcn_wmma_f32_16x16x32_bf16(false, a, false, b, (short)0, c, false, false);
}

// ---------------------------------------------------------------------------
// small helpers
// ---------------------------------------------------------------------------
__global__ void convert_f32_bf16_kernel(const float* __restrict__ src,
                                        bf16_t* __restrict__ dst, int n) {
    int i = blockIdx.x * blockDim.x + threadIdx.x;
    if (i < n) dst[i] = (bf16_t)src[i];
}

__global__ void bias_sum_kernel(const float* __restrict__ a,
                                const float* __restrict__ b,
                                float* __restrict__ out, int n) {
    int i = blockIdx.x * blockDim.x + threadIdx.x;
    if (i < n) out[i] = a[i] + b[i];
}

// ---------------------------------------------------------------------------
// Scatter-average: word_ids sorted per batch row -> binary search per (b,t).
// grid = (B, SEQ), block = 256
// ---------------------------------------------------------------------------
__global__ __launch_bounds__(256) void avg_kernel(const float* __restrict__ emb,
                                                  const int* __restrict__ wid,
                                                  bf16_t* __restrict__ avg) {
    int b = blockIdx.x;
    int t = blockIdx.y;
    const int* w = &wid[b * SSUBQ];
    int l = 0, r = SSUBQ;
    while (l < r) { int m = (l + r) >> 1; if (w[m] < t) l = m + 1; else r = m; }
    int lo = l;
    r = SSUBQ;
    while (l < r) { int m = (l + r) >> 1; if (w[m] <= t) l = m + 1; else r = m; }
    int hi = l;
    int cnt = hi - lo;
    float inv = 1.0f / (float)(cnt > 0 ? cnt : 1);
    for (int d = threadIdx.x; d < DQ; d += blockDim.x) {
        float s = 0.0f;
        for (int j = lo; j < hi; ++j)
            s += emb[((size_t)b * SSUBQ + j) * DQ + d];
        avg[((size_t)b * SEQQ + t) * DQ + d] = (bf16_t)(s * inv);
    }
}

// ---------------------------------------------------------------------------
// GEMM: C[M,N] = A[M,K](bf16) * W[N,K](bf16)^T + bias[N]
// 128x64 tile / block (256 thr = 8 waves), wave does 32x32 (4 WMMA / k-step),
// K staged 32-wide through double-buffered LDS via async global->LDS copies.
// OUTMODE: 1 = f16 out, 2 = bf16 out.  RELU in epilogue optional.
// Requires M%128==0, N%64==0, K%32==0 (true for all uses here).
// ---------------------------------------------------------------------------
template <int OUTMODE, int RELU>
__global__ __launch_bounds__(256) void gemm_wmma_kernel(
    const bf16_t* __restrict__ A, const bf16_t* __restrict__ W,
    const float* __restrict__ bias, void* __restrict__ Cout,
    int M, int N, int K) {
    __shared__ bf16_t As[2][128][40];
    __shared__ bf16_t Bs[2][64][40];
    const int tid = threadIdx.x;
    const int m0 = blockIdx.x * 128;
    const int n0 = blockIdx.y * 64;
    const int wv = tid >> 5, lane = tid & 31;
    const int wvM = wv & 3, wvN = wv >> 2;         // 4 x 2 wave grid
    const int r = lane & 15, g = lane >> 4;
    const int aRow = tid >> 1, aCol = (tid & 1) * 16;   // A: 128x32, 16 elts/thread
    const int bRow = tid >> 2, bCol = (tid & 3) * 8;    // B: 64x32,   8 elts/thread

    const int nk = K >> 5;
    f32x8 acc00 = {}, acc01 = {}, acc10 = {}, acc11 = {};

    auto stage = [&](int k0, int buf) {
        const bf16_t* ga = &A[(size_t)(m0 + aRow) * K + k0 + aCol];
        async_b128(lds_off(&As[buf][aRow][aCol]), ga);
        async_b128(lds_off(&As[buf][aRow][aCol + 8]), ga + 8);
        async_b128(lds_off(&Bs[buf][bRow][bCol]),
                   &W[(size_t)(n0 + bRow) * K + k0 + bCol]);
    };

    stage(0, 0);
    wait_async0();
    __syncthreads();
    for (int ik = 0; ik < nk; ++ik) {
        const int buf = ik & 1;
        if (ik + 1 < nk) stage((ik + 1) << 5, buf ^ 1);
        bf16x16 a0 = load_frag(&As[buf][32 * wvM + r][8 * g]);
        bf16x16 a1 = load_frag(&As[buf][32 * wvM + 16 + r][8 * g]);
        bf16x16 b0 = load_frag(&Bs[buf][32 * wvN + r][8 * g]);
        bf16x16 b1 = load_frag(&Bs[buf][32 * wvN + 16 + r][8 * g]);
        acc00 = bwmma(a0, b0, acc00);
        acc01 = bwmma(a0, b1, acc01);
        acc10 = bwmma(a1, b0, acc10);
        acc11 = bwmma(a1, b1, acc11);
        wait_async0();
        __syncthreads();
    }
    // epilogue: C/D layout -> lane r = column, row = v + 8g
    const int colA = n0 + 32 * wvN + r;
    const int colB = colA + 16;
    const float bA = bias ? bias[colA] : 0.0f;
    const float bB = bias ? bias[colB] : 0.0f;
#pragma unroll
    for (int ms = 0; ms < 2; ++ms) {
        f32x8 a0 = ms ? acc10 : acc00;
        f32x8 a1 = ms ? acc11 : acc01;
#pragma unroll
        for (int v = 0; v < 8; ++v) {
            int row = m0 + 32 * wvM + 16 * ms + v + 8 * g;
            float x0 = a0[v] + bA;
            float x1 = a1[v] + bB;
            if (RELU) { x0 = fmaxf(x0, 0.0f); x1 = fmaxf(x1, 0.0f); }
            if (OUTMODE == 1) {
                ((f16_t*)Cout)[(size_t)row * N + colA] = (f16_t)x0;
                ((f16_t*)Cout)[(size_t)row * N + colB] = (f16_t)x1;
            } else {
                ((bf16_t*)Cout)[(size_t)row * N + colA] = (bf16_t)x0;
                ((bf16_t*)Cout)[(size_t)row * N + colB] = (bf16_t)x1;
            }
        }
    }
}

// ---------------------------------------------------------------------------
// Barrier-free LSTM recurrence, one layer, BOTH directions concurrently.
// grid = (16 strips, 2 dirs), block = 256 (8 waves).
// Block owns 16 batch rows and ALL 4H gate columns -> no cross-block deps.
// Wave w owns hidden range [32w, 32w+32); a given (b,h) has all 4 gates in
// the SAME lane of 4 accumulators -> gate math is pure register work; c-state
// lives in 16 VGPRs/lane across all 128 steps.
// h state ping-pongs between two LDS buffers -> ONE barrier per step.
// Whh streams from L2 (fully resident) as global_load_b128; an opaque ZERO
// OFFSET (not an opaque pointer -- that would break global addr-space
// inference and fall back to FLAT, which also burns DScnt/LDS bandwidth)
// is re-materialized each step to stop the compiler from hoisting+spilling
// the loop-invariant weight loads.
// ---------------------------------------------------------------------------
__global__ __launch_bounds__(256) void lstm_rec_kernel(
    const f16_t* __restrict__ xpF, const f16_t* __restrict__ xpB,     // B x SEQ x 4H
    const bf16_t* __restrict__ WhhF, const bf16_t* __restrict__ WhhB, // 4H x H
    bf16_t* __restrict__ hsOut) {                                     // B x SEQ x 2H
    __shared__ bf16_t hbuf[2][16][264];
    const int tid = threadIdx.x;
    const int m0 = blockIdx.x * 16;
    const int rev = blockIdx.y;
    const f16_t* xp = rev ? xpB : xpF;
    const bf16_t* Whh = rev ? WhhB : WhhF;
    const int hOff = rev ? HQ : 0;
    const int wv = tid >> 5, lane = tid & 31;
    const int r = lane & 15, g = lane >> 4;
    const int hbase = wv * 32;

    for (int i = tid; i < 2 * 16 * 264; i += 256) ((bf16_t*)hbuf)[i] = (bf16_t)0.0f;
    float cr[16];
#pragma unroll
    for (int i = 0; i < 16; ++i) cr[i] = 0.0f;
    __syncthreads();

    for (int s = 0; s < SEQQ; ++s) {
        const int rb = s & 1, wb = rb ^ 1;
        const int tt = rev ? (SEQQ - 1 - s) : s;

        // prefetch next step's xp rows (streamed once from HBM)
        if (s + 1 < SEQQ) {
            const int tn = rev ? (SEQQ - 2 - s) : (s + 1);
            const f16_t* pf = &xp[((size_t)(m0 + (tid >> 4)) * SEQQ + tn) * GQ + (tid & 15) * 64];
            __builtin_prefetch(pf, 0, 1);
        }

        // opaque zero offset: blocks cross-step hoisting of the weight loads
        // while preserving the pointer's global address-space provenance
        uint32_t zoff = 0;
        asm volatile("" : "+s"(zoff));
        const bf16_t* Wp = Whh + zoff;

#pragma unroll
        for (int sub = 0; sub < 2; ++sub) {
            const int h = hbase + 16 * sub + r;
            const bf16_t* bbase0 = &Wp[((size_t)(0 * HQ + h)) * HQ + 8 * g];
            const bf16_t* bbase1 = &Wp[((size_t)(1 * HQ + h)) * HQ + 8 * g];
            const bf16_t* bbase2 = &Wp[((size_t)(2 * HQ + h)) * HQ + 8 * g];
            const bf16_t* bbase3 = &Wp[((size_t)(3 * HQ + h)) * HQ + 8 * g];
            f32x8 acc0 = {}, acc1 = {}, acc2 = {}, acc3 = {};
#pragma unroll
            for (int kk = 0; kk < 8; ++kk) {
                bf16x16 a = load_frag(&hbuf[rb][r][32 * kk + 8 * g]);
                acc0 = bwmma(a, load_frag(bbase0 + 32 * kk), acc0);
                acc1 = bwmma(a, load_frag(bbase1 + 32 * kk), acc1);
                acc2 = bwmma(a, load_frag(bbase2 + 32 * kk), acc2);
                acc3 = bwmma(a, load_frag(bbase3 + 32 * kk), acc3);
            }
            // gates + state update (register-local per lane)
#pragma unroll
            for (int v = 0; v < 8; ++v) {
                const int row = v + 8 * g;
                const f16_t* xr = &xp[((size_t)(m0 + row) * SEQQ + tt) * GQ];
                float gi = acc0[v] + (float)xr[h];
                float gf = acc1[v] + (float)xr[HQ + h];
                float gg = acc2[v] + (float)xr[2 * HQ + h];
                float go = acc3[v] + (float)xr[3 * HQ + h];
                float c = sigf(gf) * cr[sub * 8 + v] + sigf(gi) * tanhf(gg);
                float hv = sigf(go) * tanhf(c);
                cr[sub * 8 + v] = c;
                hbuf[wb][row][h] = (bf16_t)hv;
                hsOut[((size_t)(m0 + row) * SEQQ + tt) * (2 * HQ) + hOff + h] = (bf16_t)hv;
            }
        }
        __syncthreads();   // step s writes visible before step s+1 reads
    }
}

// ---------------------------------------------------------------------------
// logits: out[b, c, t] = dot(ff[b,t,:512], w_sbd[c,:]) + b_sbd[c]
// ---------------------------------------------------------------------------
__global__ __launch_bounds__(256) void logits_kernel(const bf16_t* __restrict__ ff,
                                                     const float* __restrict__ wsbd,
                                                     const float* __restrict__ bsbd,
                                                     float* __restrict__ out) {
    int idx = blockIdx.x * blockDim.x + threadIdx.x;   // over B*SEQ*2
    if (idx >= BQ * SEQQ * 2) return;
    int c = idx & 1;
    int bt = idx >> 1;
    int b = bt / SEQQ, t = bt % SEQQ;
    const bf16_t* f = &ff[(size_t)bt * 512];
    const float* w = &wsbd[c * 512];
    float s = bsbd[c];
    for (int k = 0; k < 512; ++k) s += (float)f[k] * w[k];
    out[((size_t)b * 2 + c) * SEQQ + t] = s;
}

// ---------------------------------------------------------------------------
// launcher
// ---------------------------------------------------------------------------
extern "C" void kernel_launch(void* const* d_in, const int* in_sizes, int n_in,
                              void* d_out, int out_size, void* d_ws, size_t ws_size,
                              hipStream_t stream) {
    (void)in_sizes; (void)n_in; (void)out_size; (void)ws_size;
    const float* emb   = (const float*)d_in[0];
    const int*   wid   = (const int*)d_in[1];
    const float* w_ih[4] = {(const float*)d_in[2], (const float*)d_in[6], (const float*)d_in[10], (const float*)d_in[14]};
    const float* w_hh[4] = {(const float*)d_in[3], (const float*)d_in[7], (const float*)d_in[11], (const float*)d_in[15]};
    const float* b_ih[4] = {(const float*)d_in[4], (const float*)d_in[8], (const float*)d_in[12], (const float*)d_in[16]};
    const float* b_hh[4] = {(const float*)d_in[5], (const float*)d_in[9], (const float*)d_in[13], (const float*)d_in[17]};
    const float* w_ff  = (const float*)d_in[18];
    const float* b_ff  = (const float*)d_in[19];
    const float* w_sbd = (const float*)d_in[20];
    const float* b_sbd = (const float*)d_in[21];
    float* out = (float*)d_out;

    // workspace carve-up (256B aligned)
    uint8_t* ws = (uint8_t*)d_ws;
    size_t off = 0;
    auto alloc = [&](size_t bytes) -> uint8_t* {
        uint8_t* p = ws + off;
        off = (off + bytes + 255) & ~(size_t)255;
        return p;
    };
    bf16_t* avg_bf = (bf16_t*)alloc((size_t)BQ * SEQQ * DQ * 2);
    bf16_t* hs_l0  = (bf16_t*)alloc((size_t)BQ * SEQQ * 512 * 2);
    bf16_t* hs_l1  = (bf16_t*)alloc((size_t)BQ * SEQQ * 512 * 2);
    bf16_t* ff_bf  = (bf16_t*)alloc((size_t)BQ * SEQQ * 512 * 2);
    f16_t*  xpF    = (f16_t*)alloc((size_t)MQ * GQ * 2);
    f16_t*  xpB    = (f16_t*)alloc((size_t)MQ * GQ * 2);
    const int wihK[4] = {DQ, DQ, 512, 512};
    bf16_t* wih_bf[4], *whh_bf[4];
    float*  bsum[4];
    for (int d = 0; d < 4; ++d) {
        wih_bf[d] = (bf16_t*)alloc((size_t)GQ * wihK[d] * 2);
        whh_bf[d] = (bf16_t*)alloc((size_t)GQ * HQ * 2);
        bsum[d]   = (float*)alloc((size_t)GQ * 4);
    }
    bf16_t* wff_bf = (bf16_t*)alloc((size_t)512 * 512 * 2);

    // 1) convert weights to bf16, sum biases
    for (int d = 0; d < 4; ++d) {
        int n1 = GQ * wihK[d];
        convert_f32_bf16_kernel<<<(n1 + 255) / 256, 256, 0, stream>>>(w_ih[d], wih_bf[d], n1);
        int n2 = GQ * HQ;
        convert_f32_bf16_kernel<<<(n2 + 255) / 256, 256, 0, stream>>>(w_hh[d], whh_bf[d], n2);
        bias_sum_kernel<<<(GQ + 255) / 256, 256, 0, stream>>>(b_ih[d], b_hh[d], bsum[d], GQ);
    }
    convert_f32_bf16_kernel<<<(512 * 512 + 255) / 256, 256, 0, stream>>>(w_ff, wff_bf, 512 * 512);

    // 2) scatter-average
    avg_kernel<<<dim3(BQ, SEQQ), 256, 0, stream>>>(emb, wid, avg_bf);

    // 3) layer 0: two x-projections, then both directions' recurrences at once
    gemm_wmma_kernel<1, 0><<<dim3(MQ / 128, GQ / 64), 256, 0, stream>>>(
        avg_bf, wih_bf[0], bsum[0], xpF, MQ, GQ, DQ);
    gemm_wmma_kernel<1, 0><<<dim3(MQ / 128, GQ / 64), 256, 0, stream>>>(
        avg_bf, wih_bf[1], bsum[1], xpB, MQ, GQ, DQ);
    lstm_rec_kernel<<<dim3(16, 2), 256, 0, stream>>>(xpF, xpB, whh_bf[0], whh_bf[1], hs_l0);

    // 4) layer 1
    gemm_wmma_kernel<1, 0><<<dim3(MQ / 128, GQ / 64), 256, 0, stream>>>(
        hs_l0, wih_bf[2], bsum[2], xpF, MQ, GQ, 512);
    gemm_wmma_kernel<1, 0><<<dim3(MQ / 128, GQ / 64), 256, 0, stream>>>(
        hs_l0, wih_bf[3], bsum[3], xpB, MQ, GQ, 512);
    lstm_rec_kernel<<<dim3(16, 2), 256, 0, stream>>>(xpF, xpB, whh_bf[2], whh_bf[3], hs_l1);

    // 5) FF (relu) then logits
    gemm_wmma_kernel<2, 1><<<dim3(MQ / 128, 512 / 64), 256, 0, stream>>>(
        hs_l1, wff_bf, b_ff, ff_bf, MQ, 512, 512);
    logits_kernel<<<(BQ * SEQQ * 2 + 255) / 256, 256, 0, stream>>>(ff_bf, w_sbd, b_sbd, out);
}
